// CrossAttention_50199577756109
// MI455X (gfx1250) — matrix operand order repaired and checked by reading, not compiled, not run on hardware
//
#include <hip/hip_runtime.h>
#include <stdint.h>

// Problem dims (all compile-time)
#define NROW 8192
#define DIN  1024
#define DKQ  512
#define DV   512

typedef __attribute__((ext_vector_type(16))) __bf16 v16bf;
typedef __attribute__((ext_vector_type(8)))  float  v8f;

union FragU { uint4 u[2]; v16bf v; };

// Native bf16 converts: backend lowers fptrunc/fpext to v_cvt_*_bf16 ops
// (and packs adjacent converts into v_cvt_pk_bf16_f32).
__device__ __forceinline__ uint16_t f32_to_bf16(float f) {
  union { __bf16 h; uint16_t u; } c;
  c.h = (__bf16)f;
  return c.u;
}
__device__ __forceinline__ float bf16_to_f32(uint16_t u) {
  union { uint16_t u; __bf16 h; } c;
  c.u = u;
  return (float)c.h;
}

// Hardware tanh if available (gfx1250 TRANS op), fallback to libm.
#if __has_builtin(__builtin_amdgcn_tanhf)
#define FAST_TANH __builtin_amdgcn_tanhf
#elif __has_builtin(__builtin_amdgcn_tanh_f32)
#define FAST_TANH __builtin_amdgcn_tanh_f32
#else
#define FAST_TANH tanhf
#endif

// A-operand fragment (M x K = 16 x 32) from a row-major tile, leading dim ld.
// ISA layout: lanes 0-15 row M=lane, K={0..7,16..23}; lanes 16-31 K={8..15,24..31}.
__device__ __forceinline__ v16bf load_frag_a(const uint16_t* base, int ld) {
  const int lane = threadIdx.x & 31;
  const uint16_t* p = base + (lane & 15) * ld + ((lane >> 4) ? 8 : 0);
  FragU f;
  f.u[0] = *(const uint4*)(p);       // K = kb .. kb+7
  f.u[1] = *(const uint4*)(p + 16);  // K = kb+16 .. kb+23
  return f.v;
}

// B-operand fragment (K x N = 32 x 16) given B^T row-major (N x K), leading dim ld.
// ISA layout: lanes 0-15 K={0..7,16..23}; lanes 16-31 K={8..15,24..31}; n = lane%16.
__device__ __forceinline__ v16bf load_frag_b(const uint16_t* base, int ld) {
  const int lane = threadIdx.x & 31;
  const uint16_t* p = base + (lane & 15) * ld + ((lane >> 4) ? 8 : 0);
  FragU f;
  f.u[0] = *(const uint4*)(p);       // K = kb .. kb+7
  f.u[1] = *(const uint4*)(p + 16);  // K = kb+16 .. kb+23
  return f.v;
}

__device__ __forceinline__ v8f wmma_bf16(v16bf a, v16bf b, v8f c) {
  return __builtin_amdgcn_wmma_f32_16x16x32_bf16(false, a, false, b,
                                                 (short)0, c, false, false);
}

// ---------------------------------------------------------------- convert
// float2 -> two bf16 packed in u32 (element counts here are all even).
__global__ __launch_bounds__(256) void cvt_f32_to_bf16(
    const float2* __restrict__ src, uint32_t* __restrict__ dst, int n2) {
  int i = blockIdx.x * blockDim.x + threadIdx.x;
  const int stride = gridDim.x * blockDim.x;
  for (; i < n2; i += stride) {
    const float2 v = src[i];
    dst[i] = (uint32_t)f32_to_bf16(v.x) | ((uint32_t)f32_to_bf16(v.y) << 16);
  }
}

// ---------------------------------------------------------------- projection
// C[NROW x 512] = A[NROW x K] @ W[512 x K]^T + bias.  C bf16; optionally also
// write C^T[512 x NROW] (for v, feeding the PV GEMM's B-operand loads).
// Block: 256 thr = 8 waves in 4x2; per-wave 32x64 output (2x4 WMMA frags).
__global__ __launch_bounds__(256) void proj_gemm(
    const uint16_t* __restrict__ A, const uint16_t* __restrict__ W,
    const float* __restrict__ bias, uint16_t* __restrict__ C,
    uint16_t* __restrict__ CT, int K)
{
  const int lane = threadIdx.x & 31;
  const int wave = threadIdx.x >> 5;
  const int wm = wave & 3;                 // 0..3
  const int wn = wave >> 2;                // 0..1
  const int row0 = blockIdx.x * 128 + wm * 32;
  const int col0 = blockIdx.y * 128 + wn * 64;

  v8f acc[2][4];
  #pragma unroll
  for (int i = 0; i < 2; ++i)
    #pragma unroll
    for (int j = 0; j < 4; ++j) acc[i][j] = (v8f){0,0,0,0,0,0,0,0};

  for (int ks = 0; ks < K; ks += 32) {
    v16bf af[2], bf[4];
    #pragma unroll
    for (int i = 0; i < 2; ++i)
      af[i] = load_frag_a(A + (size_t)(row0 + i * 16) * K + ks, K);
    #pragma unroll
    for (int j = 0; j < 4; ++j)
      bf[j] = load_frag_b(W + (size_t)(col0 + j * 16) * K + ks, K);
    #pragma unroll
    for (int i = 0; i < 2; ++i)
      #pragma unroll
      for (int j = 0; j < 4; ++j)
        acc[i][j] = wmma_bf16(af[i], bf[j], acc[i][j]);
  }

  const int lr = lane & 15;
  const int lh = lane >> 4;
  #pragma unroll
  for (int i = 0; i < 2; ++i) {
    #pragma unroll
    for (int j = 0; j < 4; ++j) {
      const int n = col0 + j * 16 + lr;
      const float bv = bias[n];
      uint16_t hv[8];
      #pragma unroll
      for (int r = 0; r < 8; ++r) {
        const int m = row0 + i * 16 + 8 * lh + r;
        const uint16_t h = f32_to_bf16(acc[i][j][r] + bv);
        C[(size_t)m * 512 + n] = h;
        hv[r] = h;
      }
      if (CT) {  // transposed store: 8 consecutive rows per lane -> b128
        const int mbase = row0 + i * 16 + 8 * lh;
        uint4 pk;
        pk.x = (uint32_t)hv[0] | ((uint32_t)hv[1] << 16);
        pk.y = (uint32_t)hv[2] | ((uint32_t)hv[3] << 16);
        pk.z = (uint32_t)hv[4] | ((uint32_t)hv[5] << 16);
        pk.w = (uint32_t)hv[6] | ((uint32_t)hv[7] << 16);
        *(uint4*)(CT + (size_t)n * NROW + mbase) = pk;
      }
    }
  }
}

// ---------------------------------------------------------------- colsum(v)
__global__ __launch_bounds__(256) void colsum_kernel(
    const uint16_t* __restrict__ V, float* __restrict__ colsum) {
  const int d = blockIdx.x * blockDim.x + threadIdx.x;
  if (d >= DV) return;
  float s = 0.0f;
  #pragma unroll 8
  for (int i = 0; i < NROW; ++i) s += bf16_to_f32(V[(size_t)i * DV + d]);
  colsum[d] = s;
}

// ---------------------------------------------------------------- attention
// Per block: 64 rows. Stream 64-col tiles: S = q@k^T (bf16 WMMA, K=512),
// P = exp(tanh(S)/sqrt(512)) -> LDS; denom += rowsum(P); acc += P @ vT-slab.
// Epilogue: out = (colsum_v - acc/denom) / (N-1).
#define MB 64
#define NB 64
#define LDP 72
__global__ __launch_bounds__(256) void attn_kernel(
    const uint16_t* __restrict__ Q, const uint16_t* __restrict__ Kb,
    const uint16_t* __restrict__ VT, const float* __restrict__ colsum,
    float* __restrict__ out)
{
  __shared__ uint16_t Pl[MB * LDP];
  __shared__ float denom[MB];

  const int lane = threadIdx.x & 31;
  const int wave = threadIdx.x >> 5;       // 0..7
  const int lr = lane & 15;
  const int lh = lane >> 4;
  const int row0 = blockIdx.x * MB;
  const int sm = wave >> 1;                // 0..3 : S-phase row block
  const int sn = wave & 1;                 // 0..1 : S-phase col half
  const int c0 = wave * 64;                // PV-phase output column slab

  if (threadIdx.x < MB) denom[threadIdx.x] = 0.0f;
  __syncthreads();

  v8f acc[4][4];
  #pragma unroll
  for (int i = 0; i < 4; ++i)
    #pragma unroll
    for (int j = 0; j < 4; ++j) acc[i][j] = (v8f){0,0,0,0,0,0,0,0};

  const float sc = 0.04419417382415922f;   // 1/sqrt(512)

  for (int jt = 0; jt < NROW; jt += NB) {
    // Prefetch next tile's k rows (64 rows x 1KB) into near caches.
    if (jt + NB < NROW) {
      const uint16_t* pf = Kb + (size_t)(jt + NB + (threadIdx.x >> 2)) * DKQ
                              + (threadIdx.x & 3) * 128;
      __builtin_prefetch(pf, 0, 1);
    }

    // ---- S = q @ k^T (64x64), waves 4x2 each 16x32 ----
    v8f s[2];
    s[0] = (v8f){0,0,0,0,0,0,0,0};
    s[1] = (v8f){0,0,0,0,0,0,0,0};
    for (int ks = 0; ks < DKQ; ks += 32) {
      v16bf aq = load_frag_a(Q + (size_t)(row0 + sm * 16) * DKQ + ks, DKQ);
      #pragma unroll
      for (int f = 0; f < 2; ++f) {
        v16bf bk = load_frag_b(Kb + (size_t)(jt + sn * 32 + f * 16) * DKQ + ks, DKQ);
        s[f] = wmma_bf16(aq, bk, s[f]);
      }
    }
    // P = exp(tanh(S)*sc) -> LDS (bf16)
    #pragma unroll
    for (int f = 0; f < 2; ++f) {
      const int n = sn * 32 + f * 16 + lr;
      #pragma unroll
      for (int r = 0; r < 8; ++r) {
        const int m = sm * 16 + 8 * lh + r;
        const float p = __expf(FAST_TANH(s[f][r]) * sc);
        Pl[m * LDP + n] = f32_to_bf16(p);
      }
    }
    __syncthreads();

    // denominators (one thread per row; fixed order -> deterministic)
    if (threadIdx.x < MB) {
      float ds = 0.0f;
      #pragma unroll 8
      for (int j = 0; j < NB; ++j) ds += bf16_to_f32(Pl[threadIdx.x * LDP + j]);
      denom[threadIdx.x] += ds;
    }

    // ---- acc += P(64x64) @ vT-slab(64 x 64-cols) ----
    #pragma unroll
    for (int ks2 = 0; ks2 < 2; ++ks2) {
      v16bf ap[4], bv[4];
      #pragma unroll
      for (int mb = 0; mb < 4; ++mb)
        ap[mb] = load_frag_a(Pl + mb * 16 * LDP + ks2 * 32, LDP);
      #pragma unroll
      for (int nb = 0; nb < 4; ++nb)
        bv[nb] = load_frag_b(VT + (size_t)(c0 + nb * 16) * NROW + jt + ks2 * 32, NROW);
      #pragma unroll
      for (int mb = 0; mb < 4; ++mb)
        #pragma unroll
        for (int nb = 0; nb < 4; ++nb)
          acc[mb][nb] = wmma_bf16(ap[mb], bv[nb], acc[mb][nb]);
    }
    __syncthreads();
  }

  // per-row reciprocal once (replaces 128 divides/thread with multiplies)
  if (threadIdx.x < MB) denom[threadIdx.x] = 1.0f / denom[threadIdx.x];
  __syncthreads();

  // ---- epilogue: out = (colsum_v - num * rdenom) / (N-1) ----
  const float inv_nm1 = 1.0f / (float)(NROW - 1);
  #pragma unroll
  for (int mb = 0; mb < 4; ++mb) {
    #pragma unroll
    for (int nb = 0; nb < 4; ++nb) {
      const int d = c0 + nb * 16 + lr;
      const float cs = colsum[d];
      #pragma unroll
      for (int r = 0; r < 8; ++r) {
        const int m = mb * 16 + 8 * lh + r;
        out[(size_t)(row0 + m) * DV + d] = (cs - acc[mb][nb][r] * denom[m]) * inv_nm1;
      }
    }
  }
}

// ---------------------------------------------------------------- launch
extern "C" void kernel_launch(void* const* d_in, const int* in_sizes, int n_in,
                              void* d_out, int out_size, void* d_ws, size_t ws_size,
                              hipStream_t stream) {
  const float* x1 = (const float*)d_in[0];
  const float* x2 = (const float*)d_in[1];
  const float* Wq = (const float*)d_in[2];
  const float* bq = (const float*)d_in[3];
  const float* Wk = (const float*)d_in[4];
  const float* bk = (const float*)d_in[5];
  const float* Wv = (const float*)d_in[6];
  const float* bv = (const float*)d_in[7];
  float* out = (float*)d_out;

  char* ws = (char*)d_ws;
  size_t off = 0;
  auto alloc = [&](size_t bytes) -> char* {
    char* p = ws + off;
    off += (bytes + 255) & ~(size_t)255;
    return p;
  };
  uint16_t* x1b = (uint16_t*)alloc((size_t)NROW * DIN * 2);
  uint16_t* x2b = (uint16_t*)alloc((size_t)NROW * DIN * 2);
  uint16_t* wqb = (uint16_t*)alloc((size_t)DKQ * DIN * 2);
  uint16_t* wkb = (uint16_t*)alloc((size_t)DKQ * DIN * 2);
  uint16_t* wvb = (uint16_t*)alloc((size_t)DV * DIN * 2);
  uint16_t* qb  = (uint16_t*)alloc((size_t)NROW * DKQ * 2);
  uint16_t* kb2 = (uint16_t*)alloc((size_t)NROW * DKQ * 2);
  uint16_t* vb  = (uint16_t*)alloc((size_t)NROW * DV * 2);
  uint16_t* vTb = (uint16_t*)alloc((size_t)DV * NROW * 2);
  float*    csum = (float*)alloc((size_t)DV * 4);

  cvt_f32_to_bf16<<<2048, 256, 0, stream>>>((const float2*)x1, (uint32_t*)x1b,
                                            NROW * DIN / 2);
  cvt_f32_to_bf16<<<2048, 256, 0, stream>>>((const float2*)x2, (uint32_t*)x2b,
                                            NROW * DIN / 2);
  cvt_f32_to_bf16<<<512, 256, 0, stream>>>((const float2*)Wq, (uint32_t*)wqb,
                                           DKQ * DIN / 2);
  cvt_f32_to_bf16<<<512, 256, 0, stream>>>((const float2*)Wk, (uint32_t*)wkb,
                                           DKQ * DIN / 2);
  cvt_f32_to_bf16<<<512, 256, 0, stream>>>((const float2*)Wv, (uint32_t*)wvb,
                                           DV * DIN / 2);

  dim3 pgrid(NROW / 128, 512 / 128);
  proj_gemm<<<pgrid, 256, 0, stream>>>(x1b, wqb, bq, qb, nullptr, DIN);
  proj_gemm<<<pgrid, 256, 0, stream>>>(x2b, wkb, bk, kb2, nullptr, DIN);
  proj_gemm<<<pgrid, 256, 0, stream>>>(x2b, wvb, bv, vb, vTb, DIN);

  colsum_kernel<<<(DV + 255) / 256, 256, 0, stream>>>(vb, csum);

  attn_kernel<<<NROW / MB, 256, 0, stream>>>(qb, kb2, vTb, csum, out);
}
